// CrossEncoder_627065225779
// MI455X (gfx1250) — matrix operand order repaired and checked by reading, not compiled, not run on hardware
//
#include <hip/hip_runtime.h>
#include <hip/hip_bf16.h>
#include <math.h>

typedef __attribute__((ext_vector_type(16))) __bf16 v16bf;
typedef __attribute__((ext_vector_type(8)))  float  v8f;

#define WIN       128
#define T_CTX     512
#define M_MENT    64
#define C_CAND    30
#define DOUT      300
#define DM        768
#define KPAD      320      // K padded: 10 slices of 32
#define NPAD      304      // N padded: 19 tiles of 16
#define KP        328      // LDS row stride (u16) -> 656B, bank-conflict free, 16B aligned
#define NT        19
#define KSLICES   10
#define TILES_PER_M 68     // 8 window tiles + 30*2 desc tiles
#define TOTAL_TILES (M_MENT * TILES_PER_M)   // 4352
#define WPB       4        // waves per block
#define NBLK      136
#define TOTAL_WAVES (NBLK * WPB)             // 544
#define ITERS     (TOTAL_TILES / TOTAL_WAVES) // 8 exactly

__device__ __forceinline__ unsigned short f2bf(float f) {
    unsigned u = __float_as_uint(f);
    unsigned r = u + 0x7FFFu + ((u >> 16) & 1u);   // RNE
    return (unsigned short)(r >> 16);
}
__device__ __forceinline__ int imax(int a, int b) { return a > b ? a : b; }
__device__ __forceinline__ int imin(int a, int b) { return a < b ? a : b; }

// ---- prep: W_ce (300x300, [k][n]) -> bf16 W^T [n=0..303][k=0..319] ----
__global__ void k_prep_wt(const float* __restrict__ Wce, unsigned short* __restrict__ wt) {
    int i = blockIdx.x * blockDim.x + threadIdx.x;
    if (i >= NPAD * KPAD) return;
    int n = i / KPAD, k = i - n * KPAD;
    float v = (n < DOUT && k < DOUT) ? Wce[k * DOUT + n] : 0.f;
    wt[i] = f2bf(v);
}

// ---- prep: ment = mention_embeddings @ W_ds + b_ds  (64x300) ----
__global__ void k_prep_ment(const float* __restrict__ me, const float* __restrict__ Wds,
                            const float* __restrict__ bds, float* __restrict__ ment) {
    int i = blockIdx.x * blockDim.x + threadIdx.x;
    if (i >= M_MENT * DOUT) return;
    int m = i / DOUT, n = i - m * DOUT;
    float s = bds[n];
    for (int k = 0; k < DM; ++k) s += me[m * DM + k] * Wds[k * DOUT + n];
    ment[i] = s;
}

// ---- prep: h0 = relu(emb[0] @ W_ce + b); zero accumulators ----
__global__ void k_prep_misc(const float* __restrict__ emb, const float* __restrict__ Wce,
                            const float* __restrict__ bce, float* __restrict__ h0,
                            float* __restrict__ wacc, float* __restrict__ dacc) {
    int i = blockIdx.x * blockDim.x + threadIdx.x;
    if (i < DOUT) {
        float s = bce[i];
        for (int k = 0; k < DOUT; ++k) s += emb[k] * Wce[k * DOUT + i];
        h0[i] = s > 0.f ? s : 0.f;
    }
    if (i < M_MENT) wacc[i] = 0.f;
    if (i < M_MENT * C_CAND) dacc[i] = 0.f;
}

// ---- main: per 16-token tile, Y = relu(E@W+b), accumulate sum(Y * ment) ----
__global__ __launch_bounds__(128) void k_main(
    const int* __restrict__ ctx, const int* __restrict__ cdesc,
    const int* __restrict__ sstart, const int* __restrict__ slen,
    const int* __restrict__ bidx, const float* __restrict__ emb,
    const float* __restrict__ bce, const unsigned short* __restrict__ wt,
    const float* __restrict__ ment, float* __restrict__ wacc, float* __restrict__ dacc) {

    __shared__ __align__(16) unsigned short lds[(WPB + 1) * 16 * KP]; // 52,480 B
    unsigned short* Wslab = lds;                         // 16 x KP (one N-tile of W^T)
    const int lane  = threadIdx.x & 31;
    const int wv    = threadIdx.x >> 5;
    unsigned short* Ebuf = lds + 16 * KP + wv * 16 * KP; // per-wave 16 x KP bf16
    const int gwave = blockIdx.x * WPB + wv;
    const int myrow = lane & 15;
    const bool hi   = lane >= 16;

    for (int it = 0; it < ITERS; ++it) {
        const int tile = gwave + it * TOTAL_WAVES;     // < 4352
        const int m  = tile / TILES_PER_M;
        const int tt = tile - m * TILES_PER_M;

        // token id for this lane's row (lanes 16..31 duplicate rows 0..15)
        int tok, cidx = 0;
        if (tt < 8) {
            int st = sstart[m], ln = slen[m];
            int lctx = (WIN - ln) >> 1;
            int rctx = WIN - ln - lctx;
            int lstart = imax(0, st - lctx);
            int lsup   = imax(0, rctx - (T_CTX - (st + ln)));
            int s0 = imin(imax(lstart - lsup, 0), T_CTX - WIN);
            tok = ctx[bidx[m] * T_CTX + s0 + tt * 16 + myrow];
        } else {
            int d = tt - 8;
            cidx = d >> 1;
            int j0 = (d & 1) * 16;
            tok = cdesc[(m * C_CAND + cidx) * 32 + j0 + myrow];
        }

        // stage 16 embedding rows -> LDS bf16, vectorized:
        // each row = 75 float4 (exactly 300 floats, 16B-aligned) + 5 zero-groups to K=320
        for (int idx = lane; idx < 16 * 80; idx += 32) {
            int row = idx / 80;
            int k4  = idx - row * 80;
            int t   = __shfl(tok, row, 32);
            float4 v = make_float4(0.f, 0.f, 0.f, 0.f);
            if (k4 < 75) v = ((const float4*)(emb + t * DOUT))[k4];
            uint2 p;
            p.x = (unsigned)f2bf(v.x) | ((unsigned)f2bf(v.y) << 16);
            p.y = (unsigned)f2bf(v.z) | ((unsigned)f2bf(v.w) << 16);
            *(uint2*)(Ebuf + row * KP + k4 * 4) = p;
        }
        __syncthreads();

        // hoist A-fragments (16x32 bf16 per K-slice) into VGPRs
        v16bf afrag[KSLICES];
        {
            const unsigned short* Er = Ebuf + myrow * KP;
            #pragma unroll
            for (int ks = 0; ks < KSLICES; ++ks) {
                int k0 = ks * 32 + (hi ? 8 : 0);
                union { uint4 q[2]; v16bf v; } u;
                u.q[0] = *(const uint4*)(Er + k0);       // K = k0..k0+7
                u.q[1] = *(const uint4*)(Er + k0 + 16);  // K = k0+16..k0+23
                afrag[ks] = u.v;
            }
        }

        float dotacc = 0.f;
        for (int nt = 0; nt < NT; ++nt) {
            // cooperative load of one 16-column slab of W^T into LDS (16B granules)
            {
                const uint4* src = (const uint4*)(wt + nt * 16 * KPAD); // 640 uint4
                for (int i = threadIdx.x; i < 640; i += blockDim.x) {
                    int row = i / 40;
                    int col = i - row * 40;
                    *(uint4*)(Wslab + row * KP + col * 8) = src[i];
                }
            }
            __syncthreads();

            // prefetch next slab (80B/lane stride across the block) to hide L2 latency
            if (nt + 1 < NT)
                __builtin_prefetch(wt + (nt + 1) * 16 * KPAD + threadIdx.x * 40, 0, 1);

            v8f acc = {};
            const unsigned short* Wr = Wslab + myrow * KP;  // column n = nt*16 + myrow
            #pragma unroll
            for (int ks = 0; ks < KSLICES; ++ks) {
                int k0 = ks * 32 + (hi ? 8 : 0);
                union { uint4 q[2]; v16bf v; } u;
                u.q[0] = *(const uint4*)(Wr + k0);
                u.q[1] = *(const uint4*)(Wr + k0 + 16);
                acc = __builtin_amdgcn_wmma_f32_16x16x32_bf16(
                    false, afrag[ks], false, u.v, (short)0, acc, false, false);
            }

            int n = nt * 16 + myrow;
            float bias = (n < DOUT) ? bce[n] : 0.f;
            float mv   = (n < DOUT) ? ment[m * DOUT + n] : 0.f;
            float s = 0.f;
            #pragma unroll
            for (int r = 0; r < 8; ++r) {
                float t = acc[r] + bias;
                s += (t > 0.f) ? t : 0.f;
            }
            dotacc += s * mv;
            __syncthreads();   // protect Wslab before next slab load
        }

        // full-wave reduction of the tile's contribution
        for (int off = 16; off >= 1; off >>= 1)
            dotacc += __shfl_xor(dotacc, off, 32);
        if (lane == 0) {
            if (tt < 8) atomicAdd(&wacc[m], dotacc);
            else        atomicAdd(&dacc[m * C_CAND + cidx], dotacc);
        }
    }
}

// ---- finalize: scores, mask, softmax, loss ----
__global__ void k_finalize(const float* __restrict__ wacc, const float* __restrict__ dacc,
                           const float* __restrict__ h0, const float* __restrict__ ment,
                           const float* __restrict__ cde, const float* __restrict__ cet,
                           float* __restrict__ out) {
    int m = threadIdx.x;
    __shared__ float lb[M_MENT];
    float sc[C_CAND + 1];
    if (m < M_MENT) {
        float h0dot = 0.f;
        for (int n = 0; n < DOUT; ++n) h0dot += h0[n] * ment[m * DOUT + n];
        for (int c = 0; c < C_CAND; ++c) {
            float s = (wacc[m] + h0dot + dacc[m * C_CAND + c]) * (1.f / 161.f);
            bool pad = (cde[(m * C_CAND + c) * DOUT] == 0.f);
            sc[c] = pad ? -100.f : s;
        }
        sc[C_CAND] = 0.f;
        // argmax (first max) over targets
        int tgt = 0; float bv = cet[m * (C_CAND + 1)];
        for (int i = 1; i <= C_CAND; ++i) {
            float v = cet[m * (C_CAND + 1) + i];
            if (v > bv) { bv = v; tgt = i; }
        }
        if (sc[tgt] == -100.f) tgt = C_CAND;
        float mx = sc[0];
        for (int i = 1; i <= C_CAND; ++i) mx = sc[i] > mx ? sc[i] : mx;
        float se = 0.f;
        for (int i = 0; i <= C_CAND; ++i) se += expf(sc[i] - mx);
        float lse = mx + logf(se);
        for (int i = 0; i <= C_CAND; ++i)
            out[1 + m * (C_CAND + 1) + i] = expf(sc[i] - lse);
        lb[m] = -(sc[tgt] - lse);
    }
    __syncthreads();
    if (m == 0) {
        float t = 0.f;
        for (int i = 0; i < M_MENT; ++i) t += lb[i];
        out[0] = t / (float)M_MENT;
    }
}

extern "C" void kernel_launch(void* const* d_in, const int* in_sizes, int n_in,
                              void* d_out, int out_size, void* d_ws, size_t ws_size,
                              hipStream_t stream) {
    const int*   ctx   = (const int*)d_in[0];
    const float* me    = (const float*)d_in[1];
    const int*   cdesc = (const int*)d_in[2];
    const float* cde   = (const float*)d_in[3];
    const float* cet   = (const float*)d_in[4];
    const int*   sst   = (const int*)d_in[5];
    const int*   sln   = (const int*)d_in[6];
    const int*   bix   = (const int*)d_in[7];
    const float* emb   = (const float*)d_in[8];
    const float* Wce   = (const float*)d_in[9];
    const float* bce   = (const float*)d_in[10];
    const float* Wds   = (const float*)d_in[11];
    const float* bds   = (const float*)d_in[12];
    float* out = (float*)d_out;

    // workspace layout
    unsigned short* wt = (unsigned short*)d_ws;                 // 304*320 u16 = 194,560 B
    float* ment = (float*)((char*)d_ws + NPAD * KPAD * 2);      // 64*300 f32
    float* h0   = ment + M_MENT * DOUT;                         // 300 f32
    float* wacc = h0 + DOUT;                                    // 64 f32
    float* dacc = wacc + M_MENT;                                // 1920 f32

    k_prep_wt  <<<(NPAD * KPAD + 255) / 256, 256, 0, stream>>>(Wce, wt);
    k_prep_ment<<<(M_MENT * DOUT + 255) / 256, 256, 0, stream>>>(me, Wds, bds, ment);
    k_prep_misc<<<8, 256, 0, stream>>>(emb, Wce, bce, h0, wacc, dacc);
    k_main     <<<NBLK, WPB * 32, 0, stream>>>(ctx, cdesc, sst, sln, bix, emb, bce,
                                               wt, ment, wacc, dacc);
    k_finalize <<<1, M_MENT, 0, stream>>>(wacc, dacc, h0, ment, cde, cet, out);
}
